// EQUICAT_56693568307222
// MI455X (gfx1250) — matrix operand order repaired and checked by reading, not compiled, not run on hardware
//
#include <hip/hip_runtime.h>
#include <math.h>

// ---------------------------------------------------------------------------
// MACE-style equivariant GNN layer for MI455X (gfx1250, wave32, WMMA).
// All channel-mixing GEMMs use v_wmma_f32_16x16x32_f16 (f16 in, f32 acc).
// Weights are pre-swizzled once into a contiguous-k B-operand register layout so
// each B tile is two global_load_b128s. Geometry / SiLU / tensor-product /
// scatter stay fp32.
// ---------------------------------------------------------------------------

typedef __attribute__((ext_vector_type(16))) _Float16 v16h;
typedef __attribute__((ext_vector_type(8)))  float    v8f;

#define LDS_STRIDE 72         // f16 row stride for staging tiles (avoid bank conflicts)
#define R_MAX 5.0f
#define AVG_NN 16.0f
#define INV_SQRT3 0.57735026918962576f
#define INV_SQRT2 0.70710678118654752f
#define SQRT3F 1.7320508075688772f
#define PI_F 3.14159265358979323846f

__device__ __forceinline__ v8f wmma16(v16h a, v16h b, v8f c) {
  // D = A(16x32 f16) * B(32x16 f16) + C(16x16 f32)
  return __builtin_amdgcn_wmma_f32_16x16x32_f16(
      /*neg_a=*/false, a, /*neg_b=*/false, b,
      /*c_mod=*/(short)0, c, /*reuse_a=*/false, /*reuse_b=*/false);
}

// A-matrix tile (16 rows x 32 K) from an LDS f16 buffer (row stride LDS_STRIDE).
// Contiguous k order: lanes 0-15 -> M=lane, halves 0..15 = K0..15;
//               lanes 16-31 -> M=lane-16, K16..31.
__device__ __forceinline__ v16h load_a_lds(const _Float16* base) {
  int lane = threadIdx.x & 31;
  int m  = lane & 15;
  int hk = (lane >> 4) & 1;
  const _Float16* r = base + m * LDS_STRIDE + hk * 16;
  v16h a;
#pragma unroll
  for (int j = 0; j < 8; ++j) { a[j] = r[j]; a[8 + j] = r[8 + j]; }
  return a;
}

// B-matrix tile from the pre-swizzled weight buffer: tiles stored as
// [kb][nb][lane][16 halves] contiguous -> one 32B vector load per lane.
__device__ __forceinline__ v16h load_b_pre(const _Float16* __restrict__ Wt,
                                           int nbt, int kb, int nb) {
  int lane = threadIdx.x & 31;
  return *(const v16h*)(Wt + ((size_t)(kb * nbt + nb) * 32 + lane) * 16);
}

__device__ __forceinline__ float silu(float x) {
  return x / (1.0f + __expf(-x));
}

// ---------------------------------------------------------------------------
// Kernel 0: pre-swizzle an fp32 weight matrix W[K x ncols] into f16 B-operand
// layout: per 32x16 tile, lane l holds halves j = W[kb*32 + (l>>4)*16 + j,
// nb*16 + (l&15)], zero-padded beyond K. One thread per (tile, lane).
// ---------------------------------------------------------------------------
__global__ void prep_w_kernel(const float* __restrict__ W, int K, int ncols,
                              _Float16* __restrict__ dst) {
  int KB = (K + 31) >> 5;
  int NBt = ncols >> 4;
  int total = KB * NBt * 32;
  int t = blockIdx.x * blockDim.x + threadIdx.x;
  if (t >= total) return;
  int lane = t & 31;
  int tile = t >> 5;
  int nb = tile % NBt;
  int kb = tile / NBt;
  int n  = lane & 15;
  int kg = ((lane >> 4) & 1) * 16;
  _Float16* o = dst + (size_t)t * 16;
#pragma unroll
  for (int j = 0; j < 16; ++j) {
    int k = kb * 32 + kg + j;
    o[j] = (_Float16)((k < K) ? W[k * ncols + nb * 16 + n] : 0.0f);
  }
}

// ---------------------------------------------------------------------------
// Kernel 1: per-edge geometry -> bessel*cutoff feats [E,8] and Y1 [E,3]
// ---------------------------------------------------------------------------
__global__ void geom_kernel(const float* __restrict__ pos,
                            const int* __restrict__ eidx, int E,
                            float* __restrict__ ef, float* __restrict__ Y1) {
  int e = blockIdx.x * blockDim.x + threadIdx.x;
  if (e >= E) return;
  int s = eidx[e];
  int r = eidx[E + e];
  float vx = pos[s * 3 + 0] - pos[r * 3 + 0];
  float vy = pos[s * 3 + 1] - pos[r * 3 + 1];
  float vz = pos[s * 3 + 2] - pos[r * 3 + 2];
  float rr = sqrtf(vx * vx + vy * vy + vz * vz);
  rr = fmaxf(rr, 1e-6f);
  float inv = 1.0f / rr;
  Y1[e * 3 + 0] = SQRT3F * vx * inv;
  Y1[e * 3 + 1] = SQRT3F * vy * inv;
  Y1[e * 3 + 2] = SQRT3F * vz * inv;
  // polynomial cutoff p=6: 1 - 28x^6 + 48x^7 - 21x^8, zeroed for x>=1
  float x  = rr * (1.0f / R_MAX);
  float x2 = x * x, x4 = x2 * x2;
  float x6 = x4 * x2, x7 = x6 * x, x8 = x4 * x4;
  float fc = (1.0f - 28.0f * x6 + 48.0f * x7 - 21.0f * x8) * (x < 1.0f ? 1.0f : 0.0f);
  float pref = sqrtf(2.0f / R_MAX) * inv * fc;
#pragma unroll
  for (int n = 1; n <= 8; ++n)
    ef[e * 8 + (n - 1)] = pref * __sinf((float)n * PI_F * rr * (1.0f / R_MAX));
}

// ---------------------------------------------------------------------------
// Kernel 2: node embedding (one-hot @ W == row select), h1 = 0
// ---------------------------------------------------------------------------
__global__ void embed_kernel(const int* __restrict__ species,
                             const float* __restrict__ embW, int N,
                             float* __restrict__ h0, float* __restrict__ h1) {
  int i = blockIdx.x * blockDim.x + threadIdx.x;
  if (i >= N * 64) return;
  int n = i >> 6, c = i & 63;
  h0[i] = embW[species[n] * 64 + c];
  h1[i * 3 + 0] = 0.0f;
  h1[i * 3 + 1] = 0.0f;
  h1[i * 3 + 2] = 0.0f;
}

// ---------------------------------------------------------------------------
// Kernel 3: zero edge-aggregation accumulators
// ---------------------------------------------------------------------------
__global__ void zero_kernel(float* __restrict__ M0, float* __restrict__ M1, int N) {
  int i = blockIdx.x * blockDim.x + threadIdx.x;
  if (i < N * 64)  M0[i] = 0.0f;
  if (i < N * 192) M1[i] = 0.0f;
}

// ---------------------------------------------------------------------------
// Kernel 4: lin-up: x0 = h0 @ W0, x1[:,:,d] = h1[:,:,d] @ W1  (WMMA)
// block = 128 threads = 4 waves; each wave owns 16 node rows.
// ---------------------------------------------------------------------------
__global__ __launch_bounds__(128) void node_linear_kernel(
    const float* __restrict__ h0, const float* __restrict__ h1,
    const _Float16* __restrict__ W0t, const _Float16* __restrict__ W1t, int N,
    float* __restrict__ x0, float* __restrict__ x1) {
  __shared__ _Float16 stg[4][16 * LDS_STRIDE];
  int wave = threadIdx.x >> 5, lane = threadIdx.x & 31;
  int n0 = blockIdx.x * 64 + wave * 16;
  _Float16* A = stg[wave];
  int hi = (lane >> 4) & 1, nn = lane & 15;
  bool full = (n0 + 16 <= N);
#pragma unroll
  for (int mat = 0; mat < 4; ++mat) {
    __syncthreads();  // WAR vs previous GEMM reads of A
    for (int idx = lane; idx < 16 * 64; idx += 32) {
      int row = idx >> 6, col = idx & 63;
      int n = n0 + row;
      float v = 0.0f;
      if (n < N)
        v = (mat == 0) ? h0[n * 64 + col] : h1[(n * 64 + col) * 3 + (mat - 1)];
      A[row * LDS_STRIDE + col] = (_Float16)v;
    }
    __syncthreads();
    const _Float16* Wt = (mat == 0) ? W0t : W1t;
    v16h a_lo = load_a_lds(A + 0);
    v16h a_hi = load_a_lds(A + 32);
#pragma unroll
    for (int nt = 0; nt < 4; ++nt) {
      v8f acc = {};
      acc = wmma16(a_lo, load_b_pre(Wt, 4, 0, nt), acc);
      acc = wmma16(a_hi, load_b_pre(Wt, 4, 1, nt), acc);
      int c = nt * 16 + nn;
      if (full) {
#pragma unroll
        for (int j = 0; j < 8; ++j) {
          int n = n0 + j + 8 * hi;
          if (mat == 0) x0[n * 64 + c] = acc[j];
          else          x1[(n * 64 + c) * 3 + (mat - 1)] = acc[j];
        }
      } else {
#pragma unroll
        for (int j = 0; j < 8; ++j) {
          int n = n0 + j + 8 * hi;
          if (n < N) {
            if (mat == 0) x0[n * 64 + c] = acc[j];
            else          x1[(n * 64 + c) * 3 + (mat - 1)] = acc[j];
          }
        }
      }
    }
  }
}

// ---------------------------------------------------------------------------
// Kernel 5 (fused edge kernel): per-edge radial MLP (WMMA) -> path-weight
// tiles kept in accumulator registers -> gather senders -> CG tensor product
// -> atomic scatter-add into receiver accumulators. 4 waves x 16 edges/block.
// ---------------------------------------------------------------------------
__global__ __launch_bounds__(128) void edge_kernel(
    const float* __restrict__ ef, const float* __restrict__ Y1g,
    const int* __restrict__ eidx, int E,
    const float* __restrict__ x0, const float* __restrict__ x1,
    const _Float16* __restrict__ W1t, const _Float16* __restrict__ W2t,
    const _Float16* __restrict__ W3t,
    float* __restrict__ M0, float* __restrict__ M1) {
  __shared__ _Float16 stgA[4][16 * LDS_STRIDE];
  __shared__ _Float16 stgB[4][16 * LDS_STRIDE];
  __shared__ float y1s[4][16][3];
  __shared__ int   snd[4][16];
  __shared__ int   rcv[4][16];
  int wave = threadIdx.x >> 5, lane = threadIdx.x & 31;
  int e0 = blockIdx.x * 64 + wave * 16;
  _Float16* A = stgA[wave];
  _Float16* B = stgB[wave];
  int hi = (lane >> 4) & 1, nn = lane & 15;
  bool full = (e0 + 16 <= E);

  // stage edge feats (K padded 8->32 with zeros) + per-edge metadata
  for (int idx = lane; idx < 16 * 32; idx += 32) {
    int row = idx >> 5, col = idx & 31;
    int e = e0 + row;
    float v = (col < 8 && e < E) ? ef[e * 8 + col] : 0.0f;
    A[row * LDS_STRIDE + col] = (_Float16)v;
  }
  if (lane < 16) {
    int e = e0 + lane;
    int s = 0, r = 0; float a = 0.f, b = 0.f, c = 0.f;
    if (e < E) { s = eidx[e]; r = eidx[E + e];
                 a = Y1g[e * 3]; b = Y1g[e * 3 + 1]; c = Y1g[e * 3 + 2]; }
    snd[wave][lane] = s; rcv[wave][lane] = r;
    y1s[wave][lane][0] = a; y1s[wave][lane][1] = b; y1s[wave][lane][2] = c;
  }
  __syncthreads();

  // MLP layer 1: t1 = silu([16,8] @ [8,64]) -> B    (K padded to 32)
  {
    v16h a_lo = load_a_lds(A + 0);
#pragma unroll
    for (int nt = 0; nt < 4; ++nt) {
      v8f acc = {};
      acc = wmma16(a_lo, load_b_pre(W1t, 4, 0, nt), acc);
#pragma unroll
      for (int j = 0; j < 8; ++j)
        B[(j + 8 * hi) * LDS_STRIDE + nt * 16 + nn] = (_Float16)silu(acc[j]);
    }
  }
  __syncthreads();

  // MLP layer 2: t2 = silu([16,64] @ [64,64]) -> A
  {
    v16h a_lo = load_a_lds(B + 0);
    v16h a_hi = load_a_lds(B + 32);
#pragma unroll
    for (int nt = 0; nt < 4; ++nt) {
      v8f acc = {};
      acc = wmma16(a_lo, load_b_pre(W2t, 4, 0, nt), acc);
      acc = wmma16(a_hi, load_b_pre(W2t, 4, 1, nt), acc);
#pragma unroll
      for (int j = 0; j < 8; ++j)
        A[(j + 8 * hi) * LDS_STRIDE + nt * 16 + nn] = (_Float16)silu(acc[j]);
    }
  }
  __syncthreads();

  // MLP layer 3 ([16,64] @ [64,320]) fused with tensor product + scatter.
  // Loop channel blocks of 16; hold the 5 path-weight tiles in registers.
  v16h a_lo = load_a_lds(A + 0);
  v16h a_hi = load_a_lds(A + 32);
#pragma unroll
  for (int cb = 0; cb < 4; ++cb) {
    v8f p0 = {}, p1 = {}, p2 = {}, p3 = {}, p4 = {};
    p0 = wmma16(a_lo, load_b_pre(W3t, 20, 0, 0 * 4 + cb), p0);
    p0 = wmma16(a_hi, load_b_pre(W3t, 20, 1, 0 * 4 + cb), p0);
    p1 = wmma16(a_lo, load_b_pre(W3t, 20, 0, 1 * 4 + cb), p1);
    p1 = wmma16(a_hi, load_b_pre(W3t, 20, 1, 1 * 4 + cb), p1);
    p2 = wmma16(a_lo, load_b_pre(W3t, 20, 0, 2 * 4 + cb), p2);
    p2 = wmma16(a_hi, load_b_pre(W3t, 20, 1, 2 * 4 + cb), p2);
    p3 = wmma16(a_lo, load_b_pre(W3t, 20, 0, 3 * 4 + cb), p3);
    p3 = wmma16(a_hi, load_b_pre(W3t, 20, 1, 3 * 4 + cb), p3);
    p4 = wmma16(a_lo, load_b_pre(W3t, 20, 0, 4 * 4 + cb), p4);
    p4 = wmma16(a_hi, load_b_pre(W3t, 20, 1, 4 * 4 + cb), p4);

    int c = cb * 16 + nn;
#pragma unroll
    for (int j = 0; j < 8; ++j) {
      int m = j + 8 * hi;
      int e = e0 + m;
      if (full || e < E) {
        int s = snd[wave][m], r = rcv[wave][m];
        float y0 = y1s[wave][m][0], y1 = y1s[wave][m][1], y2 = y1s[wave][m][2];
        float xs0 = x0[s * 64 + c];
        const float* xp = x1 + (size_t)(s * 64 + c) * 3;
        float v0 = xp[0], v1 = xp[1], v2 = xp[2];
        float a0 = p0[j], a1 = p1[j], a2 = p2[j], a3 = p3[j], a4 = p4[j];
        float dot = v0 * y0 + v1 * y1 + v2 * y2;
        float m0v = a0 * xs0 + a3 * dot * INV_SQRT3;
        float cr0 = v1 * y2 - v2 * y1;
        float cr1 = v2 * y0 - v0 * y2;
        float cr2 = v0 * y1 - v1 * y0;
        float m10 = a1 * xs0 * y0 + a2 * v0 + a4 * cr0 * INV_SQRT2;
        float m11 = a1 * xs0 * y1 + a2 * v1 + a4 * cr1 * INV_SQRT2;
        float m12 = a1 * xs0 * y2 + a2 * v2 + a4 * cr2 * INV_SQRT2;
        atomicAdd(&M0[r * 64 + c], m0v);
        float* mp = &M1[(size_t)(r * 64 + c) * 3];
        atomicAdd(mp + 0, m10);
        atomicAdd(mp + 1, m11);
        atomicAdd(mp + 2, m12);
      }
    }
  }
}

// ---------------------------------------------------------------------------
// Kernel 6: node update: h = (M/avg) @ linW (WMMA) -> product basis
// (elementwise in the C/D register layout) -> @ prod_lin_W (WMMA) -> h0,h1
// ---------------------------------------------------------------------------
__global__ __launch_bounds__(128) void node_update_kernel(
    const float* __restrict__ M0, const float* __restrict__ M1,
    const int* __restrict__ species,
    const _Float16* __restrict__ linW0t, const _Float16* __restrict__ linW1t,
    const float* __restrict__ prodW0, const float* __restrict__ prodW1,
    const _Float16* __restrict__ plinW0t, const _Float16* __restrict__ plinW1t,
    int N, float* __restrict__ h0, float* __restrict__ h1) {
  __shared__ _Float16 stg[4][16 * LDS_STRIDE];
  __shared__ int sps[4][16];
  int wave = threadIdx.x >> 5, lane = threadIdx.x & 31;
  int n0 = blockIdx.x * 64 + wave * 16;
  _Float16* A = stg[wave];
  int hi = (lane >> 4) & 1, nn = lane & 15;
  bool full = (n0 + 16 <= N);
  if (lane < 16) {
    int n = n0 + lane;
    sps[wave][lane] = (n < N) ? species[n] : 0;
  }

  v8f hacc[4][4];  // [mat][nt]: mat0 = h0, mat1..3 = h1 components
#pragma unroll
  for (int mat = 0; mat < 4; ++mat) {
    __syncthreads();
    for (int idx = lane; idx < 16 * 64; idx += 32) {
      int row = idx >> 6, col = idx & 63;
      int n = n0 + row;
      float v = 0.0f;
      if (n < N) {
        v = (mat == 0) ? M0[n * 64 + col] : M1[(n * 64 + col) * 3 + (mat - 1)];
        v *= (1.0f / AVG_NN);
      }
      A[row * LDS_STRIDE + col] = (_Float16)v;
    }
    __syncthreads();
    const _Float16* Wt = (mat == 0) ? linW0t : linW1t;
    v16h a_lo = load_a_lds(A + 0);
    v16h a_hi = load_a_lds(A + 32);
#pragma unroll
    for (int nt = 0; nt < 4; ++nt) {
      v8f acc = {};
      acc = wmma16(a_lo, load_b_pre(Wt, 4, 0, nt), acc);
      acc = wmma16(a_hi, load_b_pre(Wt, 4, 1, nt), acc);
      hacc[mat][nt] = acc;
    }
  }

  // product basis (correlation-2 symmetric contraction), in-register
#pragma unroll
  for (int nt = 0; nt < 4; ++nt) {
    int c = nt * 16 + nn;
#pragma unroll
    for (int j = 0; j < 8; ++j) {
      int m = j + 8 * hi;
      int sp = sps[wave][m];
      float hv0 = hacc[0][nt][j];
      float w1x = hacc[1][nt][j], w1y = hacc[2][nt][j], w1z = hacc[3][nt][j];
      const float* pw0 = prodW0 + (sp * 3) * 64 + c;  // [4][3][64] at layer offset
      const float* pw1 = prodW1 + (sp * 2) * 64 + c;  // [4][2][64]
      float norm2 = w1x * w1x + w1y * w1y + w1z * w1z;
      float b0 = pw0[0] * hv0 + pw0[64] * hv0 * hv0 + pw0[128] * norm2 * INV_SQRT3;
      float b1f = pw1[0] + pw1[64] * hv0;
      hacc[0][nt][j] = b0;
      hacc[1][nt][j] = b1f * w1x;
      hacc[2][nt][j] = b1f * w1y;
      hacc[3][nt][j] = b1f * w1z;
    }
  }

  // prod-lin GEMMs -> new h0, h1
#pragma unroll
  for (int mat = 0; mat < 4; ++mat) {
    __syncthreads();  // WAR vs previous GEMM reads of A
#pragma unroll
    for (int nt = 0; nt < 4; ++nt)
#pragma unroll
      for (int j = 0; j < 8; ++j)
        A[(j + 8 * hi) * LDS_STRIDE + nt * 16 + nn] = (_Float16)hacc[mat][nt][j];
    __syncthreads();
    const _Float16* Wt = (mat == 0) ? plinW0t : plinW1t;
    v16h a_lo = load_a_lds(A + 0);
    v16h a_hi = load_a_lds(A + 32);
#pragma unroll
    for (int nt = 0; nt < 4; ++nt) {
      v8f acc = {};
      acc = wmma16(a_lo, load_b_pre(Wt, 4, 0, nt), acc);
      acc = wmma16(a_hi, load_b_pre(Wt, 4, 1, nt), acc);
      int c = nt * 16 + nn;
      if (full) {
#pragma unroll
        for (int j = 0; j < 8; ++j) {
          int n = n0 + j + 8 * hi;
          if (mat == 0) h0[n * 64 + c] = acc[j];
          else          h1[(n * 64 + c) * 3 + (mat - 1)] = acc[j];
        }
      } else {
#pragma unroll
        for (int j = 0; j < 8; ++j) {
          int n = n0 + j + 8 * hi;
          if (n < N) {
            if (mat == 0) h0[n * 64 + c] = acc[j];
            else          h1[(n * 64 + c) * 3 + (mat - 1)] = acc[j];
          }
        }
      }
    }
  }
}

// ---------------------------------------------------------------------------
// Kernel 7: pack output [N,C,4] = concat(h0[...,None], h1, axis=-1)
// ---------------------------------------------------------------------------
__global__ void output_kernel(const float* __restrict__ h0,
                              const float* __restrict__ h1, int N,
                              float* __restrict__ out) {
  int i = blockIdx.x * blockDim.x + threadIdx.x;
  if (i >= N * 64) return;
  out[i * 4 + 0] = h0[i];
  out[i * 4 + 1] = h1[i * 3 + 0];
  out[i * 4 + 2] = h1[i * 3 + 1];
  out[i * 4 + 3] = h1[i * 3 + 2];
}

// ---------------------------------------------------------------------------
static void prep(const float* W, int K, int ncols, _Float16* dst, hipStream_t s) {
  int KB = (K + 31) / 32, NBt = ncols / 16;
  int total = KB * NBt * 32;
  prep_w_kernel<<<(total + 255) / 256, 256, 0, s>>>(W, K, ncols, dst);
}

extern "C" void kernel_launch(void* const* d_in, const int* in_sizes, int n_in,
                              void* d_out, int out_size, void* d_ws, size_t ws_size,
                              hipStream_t stream) {
  const float* positions  = (const float*)d_in[0];
  const int*   species    = (const int*)  d_in[1];
  const int*   eidx       = (const int*)  d_in[2];
  const float* embW       = (const float*)d_in[3];
  const float* lin_up_W0  = (const float*)d_in[4];
  const float* lin_up_W1  = (const float*)d_in[5];
  const float* mlp_W1     = (const float*)d_in[6];
  const float* mlp_W2     = (const float*)d_in[7];
  const float* mlp_W3     = (const float*)d_in[8];
  const float* lin_W0     = (const float*)d_in[9];
  const float* lin_W1     = (const float*)d_in[10];
  const float* prod_W0    = (const float*)d_in[11];
  const float* prod_W1    = (const float*)d_in[12];
  const float* plin_W0    = (const float*)d_in[13];
  const float* plin_W1    = (const float*)d_in[14];

  const int N = in_sizes[0] / 3;
  const int E = in_sizes[2] / 2;

  // workspace layout: fp32 activation buffers, then 32B-aligned f16 weights
  float* ws = (float*)d_ws;
  size_t off = 0;
  float* ef = ws + off; off += (size_t)E * 8;
  float* Y1 = ws + off; off += (size_t)E * 3;
  float* h0 = ws + off; off += (size_t)N * 64;
  float* h1 = ws + off; off += (size_t)N * 192;
  float* x0 = ws + off; off += (size_t)N * 64;
  float* x1 = ws + off; off += (size_t)N * 192;
  float* M0 = ws + off; off += (size_t)N * 64;
  float* M1 = ws + off; off += (size_t)N * 192;
  off = (off + 7) & ~(size_t)7;                 // 32B align for v16h loads
  _Float16* wb = (_Float16*)(ws + off);
  (void)ws_size; (void)n_in; (void)out_size;

  // pre-swizzled f16 weight tiles (halves): 64x64 -> 4096, 8x64 -> 2048,
  // 64x320 -> 20480
  const size_t T64 = 4096, T8 = 2048, T320 = 20480;
  _Float16 *up0t[2], *up1t[2], *m1t[2], *m2t[2], *m3t[2],
           *lin0t[2], *lin1t[2], *pl0t[2], *pl1t[2];
  size_t ho = 0;
  for (int l = 0; l < 2; ++l) {
    up0t[l]  = wb + ho; ho += T64;
    up1t[l]  = wb + ho; ho += T64;
    m1t[l]   = wb + ho; ho += T8;
    m2t[l]   = wb + ho; ho += T64;
    m3t[l]   = wb + ho; ho += T320;
    lin0t[l] = wb + ho; ho += T64;
    lin1t[l] = wb + ho; ho += T64;
    pl0t[l]  = wb + ho; ho += T64;
    pl1t[l]  = wb + ho; ho += T64;
  }
  for (int l = 0; l < 2; ++l) {
    prep(lin_up_W0 + (size_t)l * 64 * 64,  64, 64,  up0t[l],  stream);
    prep(lin_up_W1 + (size_t)l * 64 * 64,  64, 64,  up1t[l],  stream);
    prep(mlp_W1    + (size_t)l * 8 * 64,    8, 64,  m1t[l],   stream);
    prep(mlp_W2    + (size_t)l * 64 * 64,  64, 64,  m2t[l],   stream);
    prep(mlp_W3    + (size_t)l * 64 * 320, 64, 320, m3t[l],   stream);
    prep(lin_W0    + (size_t)l * 64 * 64,  64, 64,  lin0t[l], stream);
    prep(lin_W1    + (size_t)l * 64 * 64,  64, 64,  lin1t[l], stream);
    prep(plin_W0   + (size_t)l * 64 * 64,  64, 64,  pl0t[l],  stream);
    prep(plin_W1   + (size_t)l * 64 * 64,  64, 64,  pl1t[l],  stream);
  }

  geom_kernel<<<(E + 255) / 256, 256, 0, stream>>>(positions, eidx, E, ef, Y1);
  embed_kernel<<<(N * 64 + 255) / 256, 256, 0, stream>>>(species, embW, N, h0, h1);

  const int node_blocks = (N + 63) / 64;
  const int edge_blocks = (E + 63) / 64;

  for (int l = 0; l < 2; ++l) {
    node_linear_kernel<<<node_blocks, 128, 0, stream>>>(
        h0, h1, up0t[l], up1t[l], N, x0, x1);
    zero_kernel<<<(N * 192 + 255) / 256, 256, 0, stream>>>(M0, M1, N);
    edge_kernel<<<edge_blocks, 128, 0, stream>>>(
        ef, Y1, eidx, E, x0, x1, m1t[l], m2t[l], m3t[l], M0, M1);
    node_update_kernel<<<node_blocks, 128, 0, stream>>>(
        M0, M1, species, lin0t[l], lin1t[l],
        prod_W0 + (size_t)l * 4 * 3 * 64, prod_W1 + (size_t)l * 4 * 2 * 64,
        pl0t[l], pl1t[l], N, h0, h1);
  }

  output_kernel<<<(N * 64 + 255) / 256, 256, 0, stream>>>(h0, h1, N, (float*)d_out);
}
